// GptOssAttention_75557064671539
// MI455X (gfx1250) — compile-verified, matrix operand order checked by reading, not run on hardware
//
#include <hip/hip_runtime.h>

// Problem constants (match reference)
#define S_     2048
#define HID_   2880
#define NQ_    64
#define NKV_   8
#define D_     64
#define WINDOW_ 128
#define QKV_N  ((NQ_ + 2 * NKV_) * D_)   // 5120
#define NEG_   (-1e30f)

typedef __attribute__((ext_vector_type(16))) __bf16 v16bf;
typedef __attribute__((ext_vector_type(8)))  __bf16 bf16x8;
typedef __attribute__((ext_vector_type(8)))  float  v8f;

static __device__ __forceinline__ __bf16 f2bf(float f) { return (__bf16)f; }

static __device__ __forceinline__ v16bf cat8(bf16x8 lo, bf16x8 hi) {
  return __builtin_shufflevector(lo, hi, 0, 1, 2, 3, 4, 5, 6, 7,
                                         8, 9, 10, 11, 12, 13, 14, 15);
}

#define WMMA_BF16(A, B, C) \
  __builtin_amdgcn_wmma_f32_16x16x32_bf16(false, (A), false, (B), (short)0, (C), false, false)

// ---------------------------------------------------------------------------
// GEMM + bias: C[M,N] = A[M,K] @ B[K,N] + bias[N]. f32 in memory, bf16 WMMA
// with f32 accumulation, staged through LDS, register-prefetch pipelined.
// Block = 256 threads (8 waves), tile 128 x (NT*16), K-step 32.
// Requires M%128==0, N%(NT*16)==0, K%32==0. NT must be even.
// ---------------------------------------------------------------------------
#define BM 128
#define BK 32
#define PAD 8      // bf16 row padding -> 80B rows (16B-aligned, conflict-free b128)

template <int NT>
__global__ __launch_bounds__(256)
void gemm_bias_kernel(const float* __restrict__ A,
                      const float* __restrict__ Bw,
                      const float* __restrict__ bias,
                      float* __restrict__ C,
                      int M, int N, int K) {
  constexpr int BN = NT * 16;
  __shared__ __align__(16) __bf16 As[BM][BK + PAD];   // [m][k]
  __shared__ __align__(16) __bf16 Bs[BN][BK + PAD];   // transposed: [n][k]

  const int tid  = threadIdx.x;
  const int lane = tid & 31;
  const int wave = tid >> 5;
  const int row0 = blockIdx.y * BM;
  const int col0 = blockIdx.x * BN;
  const int mn   = lane & 15;          // M index (A/C) or N index (B/C)
  const int kb   = (lane >> 4) * 8;    // packed-K base per half-wave
  const int hi   = lane >> 4;
  const int wrow = wave * 16;          // wave's 16-row strip

  v8f acc[NT];
  #pragma unroll
  for (int t = 0; t < NT; ++t) acc[t] = (v8f){};

  float4 aR[4];
  float4 bR[NT / 2];

  // ---- Prologue: load tile k0=0 into registers ----
  #pragma unroll
  for (int rep = 0; rep < 4; ++rep) {
    const int idx = rep * 256 + tid;
    const int r   = idx >> 3;
    const int c4  = (idx & 7) * 4;
    aR[rep] = *(const float4*)(A + (size_t)(row0 + r) * K + c4);
  }
  #pragma unroll
  for (int rep = 0; rep < NT / 2; ++rep) {
    const int idx = rep * 256 + tid;
    const int r   = idx / (4 * NT);
    const int c4  = (idx % (4 * NT)) * 4;
    bR[rep] = *(const float4*)(Bw + (size_t)r * N + col0 + c4);
  }

  for (int k0 = 0; k0 < K; k0 += BK) {
    // ---- Store prefetched registers into LDS (convert to bf16) ----
    #pragma unroll
    for (int rep = 0; rep < 4; ++rep) {
      const int idx = rep * 256 + tid;
      const int r   = idx >> 3;
      const int c4  = (idx & 7) * 4;
      As[r][c4 + 0] = f2bf(aR[rep].x); As[r][c4 + 1] = f2bf(aR[rep].y);
      As[r][c4 + 2] = f2bf(aR[rep].z); As[r][c4 + 3] = f2bf(aR[rep].w);
    }
    #pragma unroll
    for (int rep = 0; rep < NT / 2; ++rep) {
      const int idx = rep * 256 + tid;
      const int r   = idx / (4 * NT);
      const int c4  = (idx % (4 * NT)) * 4;
      Bs[c4 + 0][r] = f2bf(bR[rep].x); Bs[c4 + 1][r] = f2bf(bR[rep].y);
      Bs[c4 + 2][r] = f2bf(bR[rep].z); Bs[c4 + 3][r] = f2bf(bR[rep].w);
    }
    __syncthreads();

    // ---- Prefetch next tile into registers (overlaps the WMMAs below) ----
    const int kn = k0 + BK;
    if (kn < K) {
      #pragma unroll
      for (int rep = 0; rep < 4; ++rep) {
        const int idx = rep * 256 + tid;
        const int r   = idx >> 3;
        const int c4  = (idx & 7) * 4;
        const float* p = A + (size_t)(row0 + r) * K + kn + c4;
        aR[rep] = *(const float4*)p;
        __builtin_prefetch(p + BK, 0, 1);     // global_prefetch one tile further
      }
      #pragma unroll
      for (int rep = 0; rep < NT / 2; ++rep) {
        const int idx = rep * 256 + tid;
        const int r   = idx / (4 * NT);
        const int c4  = (idx % (4 * NT)) * 4;
        const float* p = Bw + (size_t)(kn + r) * N + col0 + c4;
        bR[rep] = *(const float4*)p;
        __builtin_prefetch(p + (size_t)BK * N, 0, 1);
      }
    }

    // ---- Fragments from LDS (2x ds_load_b128 each), then WMMA chain ----
    const v16bf a = cat8(*(const bf16x8*)&As[wrow + mn][kb],
                         *(const bf16x8*)&As[wrow + mn][kb + 16]);
    #pragma unroll
    for (int t = 0; t < NT; t += 2) {
      const v16bf b0 = cat8(*(const bf16x8*)&Bs[t * 16 + mn][kb],
                            *(const bf16x8*)&Bs[t * 16 + mn][kb + 16]);
      const v16bf b1 = cat8(*(const bf16x8*)&Bs[(t + 1) * 16 + mn][kb],
                            *(const bf16x8*)&Bs[(t + 1) * 16 + mn][kb + 16]);
      acc[t]     = WMMA_BF16(a, b0, acc[t]);
      acc[t + 1] = WMMA_BF16(a, b1, acc[t + 1]);
    }
    __syncthreads();
  }

  // ---- Store (C layout: VGPR r -> M = r + 8*hi, N = mn) + bias ----
  #pragma unroll
  for (int t = 0; t < NT; ++t) {
    const int col = col0 + t * 16 + mn;
    const float bv = bias[col];
    #pragma unroll
    for (int r = 0; r < 8; ++r) {
      const int row = row0 + wrow + r + 8 * hi;
      C[(size_t)row * N + col] = acc[t][r] + bv;
    }
  }
}

// ---------------------------------------------------------------------------
// Rotate-half RoPE in place on q (64 heads) and k (8 heads) in the qkv buffer.
// ---------------------------------------------------------------------------
__global__ void rope_kernel(float* __restrict__ qkv,
                            const float* __restrict__ cosb,
                            const float* __restrict__ sinb) {
  int idx = blockIdx.x * blockDim.x + threadIdx.x;
  const int total = S_ * (NQ_ + NKV_) * 32;
  if (idx >= total) return;
  const int d = idx & 31;
  const int t = idx >> 5;
  const int h = t % (NQ_ + NKV_);
  const int s = t / (NQ_ + NKV_);
  float* p = qkv + (size_t)s * QKV_N + h * D_;   // q then k heads are contiguous
  const float c  = cosb[s * 32 + d];
  const float sn = sinb[s * 32 + d];
  const float x0 = p[d];
  const float x1 = p[d + 32];
  p[d]      = x0 * c - x1 * sn;
  p[d + 32] = x1 * c + x0 * sn;
}

// ---------------------------------------------------------------------------
// Sliding-window GQA attention with sink. One wave per (head, 16-query tile).
// ---------------------------------------------------------------------------
__global__ __launch_bounds__(32)
void attn_kernel(const float* __restrict__ qkv,
                 const float* __restrict__ sinks,
                 float* __restrict__ aout) {
  __shared__ __align__(16) float  sbuf[16][160];        // masked scores
  __shared__ __align__(16) __bf16 pbuf[16][168];        // exp() probs, padded rows
  __shared__ __align__(16) __bf16 Vs[64][BK + PAD];     // V chunk, transposed [d][k]
  __shared__ float rinv[16];

  const int lane = threadIdx.x & 31;
  const int h    = blockIdx.x;                 // query head
  const int qt   = blockIdx.y;                 // query tile (16 rows)
  const int hk   = h / (NQ_ / NKV_);           // kv head
  const int q0   = qt * 16;
  const int kt0  = (qt >= 8) ? (qt - 8) : 0;   // window=128 -> at most 9 key tiles
  const int ntiles = qt - kt0 + 1;
  const int NCOL   = ntiles * 16;
  const int nchunk = (NCOL + 31) / 32;

  const int mn = lane & 15;
  const int kb = (lane >> 4) * 8;
  const int hi = lane >> 4;

  // ---- Q fragments (pre-scaled by D^-0.5 = 0.125), float4 loads ----
  v16bf qa0, qa1;
  {
    const float* qp = qkv + (size_t)(q0 + mn) * QKV_N + h * D_;
    const float4 v0 = *(const float4*)(qp + kb);
    const float4 v1 = *(const float4*)(qp + kb + 4);
    const float4 v2 = *(const float4*)(qp + kb + 16);
    const float4 v3 = *(const float4*)(qp + kb + 20);
    const float4 v4 = *(const float4*)(qp + 32 + kb);
    const float4 v5 = *(const float4*)(qp + 32 + kb + 4);
    const float4 v6 = *(const float4*)(qp + 32 + kb + 16);
    const float4 v7 = *(const float4*)(qp + 32 + kb + 20);
    const float sc = 0.125f;
    qa0[0]=f2bf(v0.x*sc); qa0[1]=f2bf(v0.y*sc); qa0[2]=f2bf(v0.z*sc); qa0[3]=f2bf(v0.w*sc);
    qa0[4]=f2bf(v1.x*sc); qa0[5]=f2bf(v1.y*sc); qa0[6]=f2bf(v1.z*sc); qa0[7]=f2bf(v1.w*sc);
    qa0[8]=f2bf(v2.x*sc); qa0[9]=f2bf(v2.y*sc); qa0[10]=f2bf(v2.z*sc); qa0[11]=f2bf(v2.w*sc);
    qa0[12]=f2bf(v3.x*sc); qa0[13]=f2bf(v3.y*sc); qa0[14]=f2bf(v3.z*sc); qa0[15]=f2bf(v3.w*sc);
    qa1[0]=f2bf(v4.x*sc); qa1[1]=f2bf(v4.y*sc); qa1[2]=f2bf(v4.z*sc); qa1[3]=f2bf(v4.w*sc);
    qa1[4]=f2bf(v5.x*sc); qa1[5]=f2bf(v5.y*sc); qa1[6]=f2bf(v5.z*sc); qa1[7]=f2bf(v5.w*sc);
    qa1[8]=f2bf(v6.x*sc); qa1[9]=f2bf(v6.y*sc); qa1[10]=f2bf(v6.z*sc); qa1[11]=f2bf(v6.w*sc);
    qa1[12]=f2bf(v7.x*sc); qa1[13]=f2bf(v7.y*sc); qa1[14]=f2bf(v7.z*sc); qa1[15]=f2bf(v7.w*sc);
  }

  const float* kbase = qkv + NQ_ * D_ + hk * D_;
  const float* vbase = qkv + NQ_ * D_ + NKV_ * D_ + hk * D_;

  // ---- Phase 1: score tiles S = (Q*scale) @ K^T, masked, into LDS ----
  for (int t = 0; t < ntiles; ++t) {
    const int kt = kt0 + t;
    v8f s = {};
    {
      const float* kp = kbase + (size_t)(kt * 16 + mn) * QKV_N;
      const float4 u0 = *(const float4*)(kp + kb);
      const float4 u1 = *(const float4*)(kp + kb + 4);
      const float4 u2 = *(const float4*)(kp + kb + 16);
      const float4 u3 = *(const float4*)(kp + kb + 20);
      const float4 u4 = *(const float4*)(kp + 32 + kb);
      const float4 u5 = *(const float4*)(kp + 32 + kb + 4);
      const float4 u6 = *(const float4*)(kp + 32 + kb + 16);
      const float4 u7 = *(const float4*)(kp + 32 + kb + 20);
      v16bf b0, b1;
      b0[0]=f2bf(u0.x); b0[1]=f2bf(u0.y); b0[2]=f2bf(u0.z); b0[3]=f2bf(u0.w);
      b0[4]=f2bf(u1.x); b0[5]=f2bf(u1.y); b0[6]=f2bf(u1.z); b0[7]=f2bf(u1.w);
      b0[8]=f2bf(u2.x); b0[9]=f2bf(u2.y); b0[10]=f2bf(u2.z); b0[11]=f2bf(u2.w);
      b0[12]=f2bf(u3.x); b0[13]=f2bf(u3.y); b0[14]=f2bf(u3.z); b0[15]=f2bf(u3.w);
      b1[0]=f2bf(u4.x); b1[1]=f2bf(u4.y); b1[2]=f2bf(u4.z); b1[3]=f2bf(u4.w);
      b1[4]=f2bf(u5.x); b1[5]=f2bf(u5.y); b1[6]=f2bf(u5.z); b1[7]=f2bf(u5.w);
      b1[8]=f2bf(u6.x); b1[9]=f2bf(u6.y); b1[10]=f2bf(u6.z); b1[11]=f2bf(u6.w);
      b1[12]=f2bf(u7.x); b1[13]=f2bf(u7.y); b1[14]=f2bf(u7.z); b1[15]=f2bf(u7.w);
      s = WMMA_BF16(qa0, b0, s);
      s = WMMA_BF16(qa1, b1, s);
    }
    #pragma unroll
    for (int r = 0; r < 8; ++r) {
      const int mq = r + 8 * hi;
      const int gq = q0 + mq;
      const int gk = kt * 16 + mn;
      const bool ok = (gk <= gq) && (gq - gk < WINDOW_);
      sbuf[mq][t * 16 + mn] = ok ? s[r] : NEG_;
    }
  }
  __syncthreads();

  // ---- Phase 2: softmax per row with sink in denominator ----
  if (lane < 16) {
    const float sk = sinks[h];
    float mx = sk;
    for (int j = 0; j < NCOL; ++j) mx = fmaxf(mx, sbuf[lane][j]);
    float denom = __expf(sk - mx);
    const int padcol = nchunk * 32;
    for (int j = 0; j < padcol; ++j) {
      float e = 0.0f;
      if (j < NCOL) { e = __expf(sbuf[lane][j] - mx); denom += e; }
      pbuf[lane][j] = f2bf(e);
    }
    rinv[lane] = 1.0f / denom;
  }
  __syncthreads();

  // ---- Phase 3: O = P @ V over 32-key chunks, V staged through LDS ----
  v8f o0 = {}, o1 = {}, o2 = {}, o3 = {};
  for (int c = 0; c < nchunk; ++c) {
    // P fragment first: its ds_loads overlap the global V loads below
    const v16bf pa = cat8(*(const bf16x8*)&pbuf[mn][c * 32 + kb],
                          *(const bf16x8*)&pbuf[mn][c * 32 + kb + 16]);

    // Stage V chunk (32 keys x 64 dims), transposed to [d][k] bf16
    #pragma unroll
    for (int rep = 0; rep < 16; ++rep) {
      const int idx = rep * 32 + lane;         // 0..511
      const int r   = idx >> 4;                // key within chunk: 0..31
      const int c4  = (idx & 15) * 4;          // dim: 0,4,..,60
      int g = kt0 * 16 + c * 32 + r;
      if (g > S_ - 1) g = S_ - 1;              // padded keys have P==0
      const float4 v = *(const float4*)(vbase + (size_t)g * QKV_N + c4);
      Vs[c4 + 0][r] = f2bf(v.x); Vs[c4 + 1][r] = f2bf(v.y);
      Vs[c4 + 2][r] = f2bf(v.z); Vs[c4 + 3][r] = f2bf(v.w);
    }
    __syncthreads();

    // All V fragments issued before the WMMA chain (partial dscnt waits)
    const v16bf vb0 = cat8(*(const bf16x8*)&Vs[ 0 + mn][kb],
                           *(const bf16x8*)&Vs[ 0 + mn][kb + 16]);
    const v16bf vb1 = cat8(*(const bf16x8*)&Vs[16 + mn][kb],
                           *(const bf16x8*)&Vs[16 + mn][kb + 16]);
    const v16bf vb2 = cat8(*(const bf16x8*)&Vs[32 + mn][kb],
                           *(const bf16x8*)&Vs[32 + mn][kb + 16]);
    const v16bf vb3 = cat8(*(const bf16x8*)&Vs[48 + mn][kb],
                           *(const bf16x8*)&Vs[48 + mn][kb + 16]);
    o0 = WMMA_BF16(pa, vb0, o0);
    o1 = WMMA_BF16(pa, vb1, o1);
    o2 = WMMA_BF16(pa, vb2, o2);
    o3 = WMMA_BF16(pa, vb3, o3);
    __syncthreads();
  }

  // ---- Store: aout [S, NQ*D], normalized ----
  #pragma unroll
  for (int t = 0; t < 4; ++t) {
    const v8f o = (t == 0) ? o0 : (t == 1) ? o1 : (t == 2) ? o2 : o3;
    #pragma unroll
    for (int r = 0; r < 8; ++r) {
      const int mq = r + 8 * hi;
      aout[(size_t)(q0 + mq) * (NQ_ * D_) + h * D_ + t * 16 + mn] = o[r] * rinv[mq];
    }
  }
}

// ---------------------------------------------------------------------------
// Launcher
// ---------------------------------------------------------------------------
extern "C" void kernel_launch(void* const* d_in, const int* in_sizes, int n_in,
                              void* d_out, int out_size, void* d_ws, size_t ws_size,
                              hipStream_t stream) {
  const float* hidden = (const float*)d_in[0];
  const float* qkv_w  = (const float*)d_in[1];
  const float* qkv_b  = (const float*)d_in[2];
  const float* o_w    = (const float*)d_in[3];
  const float* o_b    = (const float*)d_in[4];
  const float* cosb   = (const float*)d_in[5];
  const float* sinb   = (const float*)d_in[6];
  const float* sinks  = (const float*)d_in[7];
  float* out = (float*)d_out;

  float* qkv  = (float*)d_ws;                       // [S, 5120] f32
  float* aout = qkv + (size_t)S_ * QKV_N;           // [S, 4096] f32

  // 1) QKV projection + bias: [2048,2880] @ [2880,5120]; 128x128 block tile
  gemm_bias_kernel<8><<<dim3(QKV_N / 128, S_ / BM), 256, 0, stream>>>(
      hidden, qkv_w, qkv_b, qkv, S_, QKV_N, HID_);

  // 2) RoPE on q and k in place
  {
    const int total = S_ * (NQ_ + NKV_) * 32;
    rope_kernel<<<(total + 255) / 256, 256, 0, stream>>>(qkv, cosb, sinb);
  }

  // 3) Sliding-window GQA attention with sink
  attn_kernel<<<dim3(NQ_, S_ / 16), 32, 0, stream>>>(qkv, sinks, aout);

  // 4) Output projection + bias: [2048,4096] @ [4096,2880]; 128x64 block tile
  gemm_bias_kernel<4><<<dim3(HID_ / 64, S_ / BM), 256, 0, stream>>>(
      aout, o_w, o_b, out, S_, HID_, NQ_ * D_);
}
